// _MS_NonLocalBlockND_16896401343292
// MI455X (gfx1250) — matrix operand order repaired and checked by reading, not compiled, not run on hardware
//
#include <hip/hip_runtime.h>
#include <math.h>

// ---------------------------------------------------------------------------
// NonLocal block (flash-attention style) for MI455X / gfx1250, wave32 + WMMA.
// bf16 WMMA operands, fp32 accumulate, fp32 softmax/BN statistics.
// K/G tiles streamed to LDS with global_load_async_to_lds_b128 (double
// buffered, ASYNCcnt), shared by all 4 waves of the block.
// ---------------------------------------------------------------------------

typedef __attribute__((ext_vector_type(16))) __bf16 v16bf;
typedef __attribute__((ext_vector_type(2)))  __bf16 v2bf;
typedef __attribute__((ext_vector_type(8)))  float  v8f;

#define B_  4
#define C_  256
#define N_  4096      // H*W
#define CP_ 64        // inter/context channels

#define WMMA_BF16(A, Bm, Cm) \
  __builtin_amdgcn_wmma_f32_16x16x32_bf16(false, (A), false, (Bm), (short)0, (Cm), false, false)

// LDS byte offset of a __shared__ object (flat aperture: addr[31:0] = LDS addr)
__device__ __forceinline__ uint32_t lds_off(const void* p) {
  return (uint32_t)(uintptr_t)p;
}

// Async global->LDS copy of 16B per lane (CDNA5 GLOBAL_LOAD_ASYNC_TO_LDS_B128,
// GV addressing, tracked by ASYNCcnt).
__device__ __forceinline__ void async_g2l_b128(uint32_t lds_byte, const void* gaddr) {
  asm volatile("global_load_async_to_lds_b128 %0, %1, off"
               :: "v"(lds_byte), "v"((uint64_t)gaddr) : "memory");
}
__device__ __forceinline__ void wait_async0() {
  asm volatile("s_wait_asynccnt 0x0" ::: "memory");
}

// ---- fragment loaders (per CDNA5 ISA 7.12.2 VGPR layouts, wave32) ----------

// A-matrix 16x32 bf16, row-major source (ld elements between rows).
__device__ __forceinline__ v16bf ldA_bf(const __bf16* __restrict__ src, int ld) {
  const int lane = threadIdx.x & 31, row = lane & 15, hi = lane >> 4;
  v16bf a;
#pragma unroll
  for (int v = 0; v < 8; ++v) {
    const int kb = ((v < 4) ? 0 : 16) + hi * 8 + (v & 3) * 2;
    a[2 * v]     = src[row * ld + kb];
    a[2 * v + 1] = src[row * ld + kb + 1];
  }
  return a;
}

// A fragment from a per-lane-linear staging buffer (each lane's 16 halves
// contiguous & 32B aligned -> 2x ds_load_b128).
__device__ __forceinline__ v16bf ldA_lin(const __bf16* src) {
  const int lane = threadIdx.x & 31;
  const __bf16* p = src + lane * 16;
  v16bf a;
#pragma unroll
  for (int i = 0; i < 16; ++i) a[i] = p[i];
  return a;
}

// Same as ldA_bf but fp32 source (weights), convert to bf16 on the fly.
__device__ __forceinline__ v16bf ldA_f32(const float* __restrict__ src, int ld) {
  const int lane = threadIdx.x & 31, row = lane & 15, hi = lane >> 4;
  v16bf a;
#pragma unroll
  for (int v = 0; v < 8; ++v) {
    const int kb = ((v < 4) ? 0 : 16) + hi * 8 + (v & 3) * 2;
    a[2 * v]     = (__bf16)src[row * ld + kb];
    a[2 * v + 1] = (__bf16)src[row * ld + kb + 1];
  }
  return a;
}

// B-matrix 32x16 where B[k][col] = src[col*ld + k]: per-lane 2x 16B chunks (k = 8*hi.., 16+8*hi..).
__device__ __forceinline__ v16bf ldBt(const __bf16* src, int ld) {
  const int lane = threadIdx.x & 31, col = lane & 15, hi = lane >> 4;
  const __bf16* p = src + (size_t)col * ld + hi * 8;
  v16bf b;
#pragma unroll
  for (int i = 0; i < 8; ++i) { b[i] = p[i]; b[i + 8] = p[16 + i]; }
  return b;
}

// ---------------------------------------------------------------------------
// Kernel 1: fused theta/phi/g projections.  out[o,n] = sum_c w[o,c]*x[b,c,n]+b
// x tile is staged TRANSPOSED in LDS (sx2[n][c], bf16, pair-packed stores) so
// every B-operand read is two 16B per-lane ds_load_b128 chunks (k = 8*hi.., 16+8*hi..).
// Q = theta as bf16 [B][N][64], Km = phi^T [B][N][64], Gt = g^T [B][64][N].
// ---------------------------------------------------------------------------
__global__ __launch_bounds__(128) void k_proj(
    const float* __restrict__ x,
    const float* __restrict__ tw, const float* __restrict__ tb,
    const float* __restrict__ pw, const float* __restrict__ pb,
    const float* __restrict__ gw, const float* __restrict__ gb,
    __bf16* __restrict__ Q, __bf16* __restrict__ Km, __bf16* __restrict__ Gt) {
  __shared__ __attribute__((aligned(32))) __bf16 sx2[64 * 32];  // [n][c]
  const int b = blockIdx.y, nb = blockIdx.x;
  const int wave = threadIdx.x >> 5, lane = threadIdx.x & 31;
  const int col = lane & 15, hi = lane >> 4;
  const int obase = wave * 16;

  v8f acc[3][4];
#pragma unroll
  for (int p = 0; p < 3; ++p)
#pragma unroll
    for (int t = 0; t < 4; ++t) acc[p][t] = (v8f){0.f,0.f,0.f,0.f,0.f,0.f,0.f,0.f};

  const float* wmat[3] = {tw, pw, gw};

  for (int c0 = 0; c0 < C_; c0 += 32) {
    // stage x[b, c0..c0+32, nb*64..+64] transposed into LDS: sx2[n][c]
    // pair-pack two adjacent channels -> one cvt_pk + one ds_store_b32
#pragma unroll
    for (int q = threadIdx.x; q < 1024; q += 128) {
      const int n = q & 63, cp = q >> 6;                  // cp: 0..15 channel pair
      const float* xp = x + ((size_t)b * C_ + c0 + 2 * cp) * (size_t)N_ + nb * 64 + n;
      v2bf pk;
      pk[0] = (__bf16)xp[0];
      pk[1] = (__bf16)xp[N_];
      *reinterpret_cast<v2bf*>(&sx2[n * 32 + 2 * cp]) = pk;
    }
    __syncthreads();

    v16bf Bx[4];
#pragma unroll
    for (int t = 0; t < 4; ++t) Bx[t] = ldBt(sx2 + (t * 16) * 32, 32);
#pragma unroll
    for (int p = 0; p < 3; ++p) {
      const v16bf Aw = ldA_f32(wmat[p] + obase * C_ + c0, C_);
#pragma unroll
      for (int t = 0; t < 4; ++t) acc[p][t] = WMMA_BF16(Aw, Bx[t], acc[p][t]);
    }
    __syncthreads();
  }

  const float* bias[3] = {tb, pb, gb};
#pragma unroll
  for (int p = 0; p < 3; ++p) {
#pragma unroll
    for (int r = 0; r < 8; ++r) {
      const int o = obase + r + 8 * hi;
      const float bv = bias[p][o];
#pragma unroll
      for (int t = 0; t < 4; ++t) {
        const int n = nb * 64 + t * 16 + col;
        const float v = acc[p][t][r] + bv;
        if (p == 0)      Q [((size_t)b * N_ + n) * CP_ + o] = (__bf16)v;
        else if (p == 1) Km[((size_t)b * N_ + n) * CP_ + o] = (__bf16)v;
        else             Gt[((size_t)b * CP_ + o) * (size_t)N_ + n] = (__bf16)v;
      }
    }
  }
}

// ---------------------------------------------------------------------------
// Kernel 2: fused attention (flash style, online softmax).
// Block = 4 waves x 16 query rows. All waves share per-step K (32x64) and
// G (64x32) tiles staged in LDS via async copies, double buffered:
//   iter i:  wait ASYNCcnt==0; barrier; issue async for step i+1; compute i.
// P tile written to LDS in A-fragment per-lane-linear order (2x ds_load_b128).
// ---------------------------------------------------------------------------
__global__ __launch_bounds__(128) void k_attn(
    const __bf16* __restrict__ Q, const __bf16* __restrict__ Km,
    const __bf16* __restrict__ Gt, __bf16* __restrict__ Y) {
  __shared__ __attribute__((aligned(32))) __bf16 sK[2][32 * 64];  // [j-local][k]
  __shared__ __attribute__((aligned(32))) __bf16 sG[2][64 * 32];  // [channel][j-local]
  __shared__ __attribute__((aligned(32))) __bf16 sp2[4][512];     // P, A-linear order

  const int b = blockIdx.y;
  const int tid  = threadIdx.x;
  const int wave = tid >> 5, lane = tid & 31;
  const int col = lane & 15, hi = lane >> 4;
  const int i0 = (blockIdx.x * 4 + wave) * 16;

  const __bf16* qrow = Q + ((size_t)b * N_ + i0) * CP_;
  const v16bf Aq0 = ldA_bf(qrow, CP_);
  const v16bf Aq1 = ldA_bf(qrow + 32, CP_);

  v8f acc[4];
#pragma unroll
  for (int t = 0; t < 4; ++t) acc[t] = (v8f){0.f,0.f,0.f,0.f,0.f,0.f,0.f,0.f};
  float mrow[8], lrow[8];
#pragma unroll
  for (int r = 0; r < 8; ++r) { mrow[r] = -3.0e38f; lrow[r] = 0.f; }

  const float scale = 0.125f;  // 64^-0.5
  const __bf16* KmB = Km + (size_t)b * N_ * CP_;
  const __bf16* GtB = Gt + (size_t)b * CP_ * (size_t)N_;

  auto prefetch = [&](int buf, int j0) {
    // K tile: 32 rows x 64 halves, contiguous 4KB -> 2x B128 per thread
#pragma unroll
    for (int c = 0; c < 2; ++c) {
      const int e = (tid + c * 128) * 8;                 // half index
      async_g2l_b128(lds_off(&sK[buf][e]), KmB + (size_t)j0 * CP_ + e);
    }
    // G tile: 64 rows x 32 halves (row stride N_) -> 2x B128 per thread
#pragma unroll
    for (int c = 0; c < 2; ++c) {
      const int chunk = tid + c * 128;                   // 0..255
      const int ch = chunk >> 2, off = (chunk & 3) * 8;  // 8 halves = 16B
      async_g2l_b128(lds_off(&sG[buf][ch * 32 + off]),
                     GtB + (size_t)ch * N_ + j0 + off);
    }
  };

  prefetch(0, 0);

  for (int j0 = 0; j0 < N_; j0 += 32) {
    const int buf = (j0 >> 5) & 1;
    wait_async0();
    __syncthreads();                       // tiles for this step visible block-wide
    if (j0 + 32 < N_) prefetch(buf ^ 1, j0 + 32);

    const __bf16* sKb = sK[buf];
    const __bf16* sGb = sG[buf];

    v8f S0 = (v8f){0.f,0.f,0.f,0.f,0.f,0.f,0.f,0.f};
    v8f S1 = (v8f){0.f,0.f,0.f,0.f,0.f,0.f,0.f,0.f};
    S0 = WMMA_BF16(Aq0, ldBt(sKb,                64), S0);
    S0 = WMMA_BF16(Aq1, ldBt(sKb + 32,           64), S0);
    S1 = WMMA_BF16(Aq0, ldBt(sKb + 16 * 64,      64), S1);
    S1 = WMMA_BF16(Aq1, ldBt(sKb + 16 * 64 + 32, 64), S1);

    // P destination slots in A-fragment linear order:
    //   element (row, kcol) -> lane (row + 16*((kcol>>3)&1));
    //   t=0 (kcol=col): slot = col&7 ; t=1 (kcol=col+16): slot += 8
    const int pbase0 = ((col >> 3) << 8) + (col & 7);

#pragma unroll
    for (int r = 0; r < 8; ++r) {
      const float s0 = S0[r] * scale, s1 = S1[r] * scale;
      float mx = fmaxf(s0, s1);
#pragma unroll
      for (int m = 1; m < 16; m <<= 1) mx = fmaxf(mx, __shfl_xor(mx, m, 32));
      const float nm   = fmaxf(mrow[r], mx);
      const float corr = __expf(mrow[r] - nm);
      const float p0 = __expf(s0 - nm), p1 = __expf(s1 - nm);
      float rs = p0 + p1;
#pragma unroll
      for (int m = 1; m < 16; m <<= 1) rs += __shfl_xor(rs, m, 32);
      lrow[r] = lrow[r] * corr + rs;
      mrow[r] = nm;
      acc[0][r] *= corr; acc[1][r] *= corr; acc[2][r] *= corr; acc[3][r] *= corr;
      const int row = r + 8 * hi;
      sp2[wave][pbase0 + row * 16]     = (__bf16)p0;
      sp2[wave][pbase0 + row * 16 + 8] = (__bf16)p1;
    }
    __syncthreads();
    const v16bf Ap = ldA_lin(sp2[wave]);
#pragma unroll
    for (int t = 0; t < 4; ++t)
      acc[t] = WMMA_BF16(Ap, ldBt(sGb + t * 16 * 32, 32), acc[t]);
    __syncthreads();
  }

#pragma unroll
  for (int r = 0; r < 8; ++r) {
    const float inv = 1.0f / lrow[r];
    const size_t row = (size_t)b * N_ + i0 + r + 8 * hi;
#pragma unroll
    for (int t = 0; t < 4; ++t)
      Y[row * CP_ + t * 16 + col] = (__bf16)(acc[t][r] * inv);
  }
}

// ---------------------------------------------------------------------------
// Kernel 3: wy[o,n] = sum_c2 W_w[o,c2]*y[b,n,c2] + W_b[o] (WMMA, K=64),
// plus per-channel sum / sum^2 atomics for BatchNorm.
// ---------------------------------------------------------------------------
__global__ __launch_bounds__(128) void k_wconv(
    const __bf16* __restrict__ Y, const float* __restrict__ Ww,
    const float* __restrict__ Wb, float* __restrict__ WY,
    float* __restrict__ stats) {
  const int b = blockIdx.z;
  const int wave = threadIdx.x >> 5, lane = threadIdx.x & 31;
  const int col = lane & 15, hi = lane >> 4;
  const int ob = blockIdx.y * 4 + wave;
  const int nb = blockIdx.x;

  const v16bf A0 = ldA_f32(Ww + ob * 16 * CP_, CP_);
  const v16bf A1 = ldA_f32(Ww + ob * 16 * CP_ + 32, CP_);

  v8f acc[4];
#pragma unroll
  for (int t = 0; t < 4; ++t) {
    acc[t] = (v8f){0.f,0.f,0.f,0.f,0.f,0.f,0.f,0.f};
    const __bf16* yp = Y + ((size_t)b * N_ + nb * 64 + t * 16) * CP_;
    acc[t] = WMMA_BF16(A0, ldBt(yp, CP_), acc[t]);
    acc[t] = WMMA_BF16(A1, ldBt(yp + 32, CP_), acc[t]);
  }

#pragma unroll
  for (int r = 0; r < 8; ++r) {
    const int o = ob * 16 + r + 8 * hi;
    const float bv = Wb[o];
    float s = 0.f, s2 = 0.f;
#pragma unroll
    for (int t = 0; t < 4; ++t) {
      const float v = acc[t][r] + bv;
      WY[((size_t)b * C_ + o) * (size_t)N_ + nb * 64 + t * 16 + col] = v;
      s += v; s2 += v * v;
    }
#pragma unroll
    for (int m = 1; m < 16; m <<= 1) { s += __shfl_xor(s, m, 32); s2 += __shfl_xor(s2, m, 32); }
    if (col == 0) { atomicAdd(&stats[o], s); atomicAdd(&stats[C_ + o], s2); }
  }
}

// ---------------------------------------------------------------------------
// Kernel 4: BatchNorm finalize (batch stats, biased variance).
// ---------------------------------------------------------------------------
__global__ __launch_bounds__(256) void k_bn(
    const float* __restrict__ WY, const float* __restrict__ stats,
    const float* __restrict__ gamma, const float* __restrict__ beta,
    float* __restrict__ out) {
  const size_t idx = (size_t)blockIdx.x * blockDim.x + threadIdx.x;
  if (idx >= (size_t)B_ * C_ * N_) return;
  const int o = (int)((idx >> 12) & (C_ - 1));
  const float invcnt = 1.0f / (float)(B_ * N_);
  const float mean = stats[o] * invcnt;
  const float var  = stats[C_ + o] * invcnt - mean * mean;
  out[idx] = gamma[o] * (WY[idx] - mean) * rsqrtf(var + 1e-5f) + beta[o];
}

// ---------------------------------------------------------------------------
extern "C" void kernel_launch(void* const* d_in, const int* in_sizes, int n_in,
                              void* d_out, int out_size, void* d_ws, size_t ws_size,
                              hipStream_t stream) {
  const float* x   = (const float*)d_in[0];
  const float* tw  = (const float*)d_in[1];
  const float* tb  = (const float*)d_in[2];
  const float* pw  = (const float*)d_in[3];
  const float* pb  = (const float*)d_in[4];
  const float* gw  = (const float*)d_in[5];
  const float* gb  = (const float*)d_in[6];
  const float* Ww  = (const float*)d_in[7];
  const float* Wb  = (const float*)d_in[8];
  const float* bng = (const float*)d_in[9];
  const float* bnb = (const float*)d_in[10];
  float* out = (float*)d_out;

  // workspace: Q 2MB | Km 2MB | Gt 2MB | Y 2MB | WY 16MB | stats 2KB
  char* ws = (char*)d_ws;
  __bf16* Q     = (__bf16*)(ws);
  __bf16* Km    = (__bf16*)(ws + (2u << 20));
  __bf16* Gt    = (__bf16*)(ws + (4u << 20));
  __bf16* Yb    = (__bf16*)(ws + (6u << 20));
  float*  WY    = (float*) (ws + (8u << 20));
  float*  stats = (float*) (ws + (24u << 20));

  hipMemsetAsync(stats, 0, 2 * C_ * sizeof(float), stream);

  k_proj <<<dim3(N_ / 64, B_),    128, 0, stream>>>(x, tw, tb, pw, pb, gw, gb, Q, Km, Gt);
  k_attn <<<dim3(N_ / 64, B_),    128, 0, stream>>>(Q, Km, Gt, Yb);
  k_wconv<<<dim3(N_ / 64, 4, B_), 128, 0, stream>>>(Yb, Ww, Wb, WY, stats);
  k_bn   <<<(B_ * C_ * N_) / 256, 256, 0, stream>>>(WY, stats, bng, bnb, out);
}
